// MLP_LSTM_Network_83528523973236
// MI455X (gfx1250) — matrix operand order repaired and checked by reading, not compiled, not run on hardware
//
#include <hip/hip_runtime.h>
#include <stdint.h>

// ---------------------------------------------------------------------------
// MI455X (gfx1250) implementation: bf16 WMMA (f32 accumulate) for every GEMM.
// MLP is two big GEMMs; the LSTM is 128 sequential {gates-GEMM, cell, outproj}
// launches (kernel boundary == step barrier; weights stay resident in L2).
// ---------------------------------------------------------------------------

typedef __attribute__((ext_vector_type(16))) __bf16 v16bf;
typedef __attribute__((ext_vector_type(8)))  __bf16 v8bf;
typedef __attribute__((ext_vector_type(8)))  float  v8f;

#define B_   256      // batch
#define S_   128      // sequence
#define IND  192      // MLP input dim
#define MH   1024     // MLP hidden
#define MO   256      // MLP out
#define HID  1024     // LSTM hidden
#define ACT  64       // action dim
#define LIN  320      // LSTM input = MO + ACT
#define KG   1344     // fused gate K = LIN + HID
#define G4   4096     // 4*HID gates

// A fragment, 16x32 bf16 (ISA 7.12.2): lane<16 holds K {0..7,16..23} of row
// (lane&15); lane>=16 holds K {8..15,24..31}. Two 16B contiguous loads.
__device__ __forceinline__ v16bf load_a_frag(const __bf16* row_ptr, int kofs, int hi) {
  union { v16bf v; v8bf h[2]; } u;
  u.h[0] = *(const v8bf*)(row_ptr + kofs + hi * 8);
  u.h[1] = *(const v8bf*)(row_ptr + kofs + 16 + hi * 8);
  return u.v;
}

// B fragment, 32x16 bf16: lane<16 holds K {0..7,16..23} of column (lane&15),
// lane>=16 holds K {8..15,24..31}.  B[k,n] = W[n,k] (torch weight layout) ->
// two 16-byte loads from row n of W.
__device__ __forceinline__ v16bf load_b_frag(const __bf16* wrow_ptr, int kofs, int khalf) {
  union { v16bf v; v8bf h[2]; } u;
  u.h[0] = *(const v8bf*)(wrow_ptr + kofs + (khalf >> 1));
  u.h[1] = *(const v8bf*)(wrow_ptr + kofs + 16 + (khalf >> 1));
  return u.v;
}

__device__ __forceinline__ v8f wmma_bf16(v16bf a, v16bf b, v8f c) {
  return __builtin_amdgcn_wmma_f32_16x16x32_bf16(false, a, false, b, (short)0, c,
                                                 false, false);
}

// ------------------------- pre-pack kernels --------------------------------

__global__ void pack_x_kernel(const float* __restrict__ obs,
                              const float* __restrict__ obst,
                              const float* __restrict__ msk,
                              __bf16* __restrict__ xbf) {
  int idx = blockIdx.x * blockDim.x + threadIdx.x;
  if (idx >= B_ * S_ * IND) return;
  int m = idx / IND, k = idx - m * IND;
  float v;
  if (k < 64) {
    v = obs[m * 64 + k];
  } else {
    int r = k - 64, o = r >> 4, d = r & 15;
    v = obst[m * 128 + o * 16 + d] * msk[m * 8 + o];
  }
  xbf[idx] = (__bf16)v;
}

__global__ void cvt_bf16_kernel(const float* __restrict__ src,
                                __bf16* __restrict__ dst, int n) {
  int i = blockIdx.x * blockDim.x + threadIdx.x;
  if (i < n) dst[i] = (__bf16)src[i];
}

// Fuse W_ih [4096,320] and W_hh [4096,1024] -> Wg [4096,1344] bf16.
__global__ void pack_wg_kernel(const float* __restrict__ wih,
                               const float* __restrict__ whh,
                               __bf16* __restrict__ wg) {
  int idx = blockIdx.x * blockDim.x + threadIdx.x;
  if (idx >= G4 * KG) return;
  int row = idx / KG, k = idx - row * KG;
  float v = (k < LIN) ? wih[row * LIN + k] : whh[row * HID + (k - LIN)];
  wg[idx] = (__bf16)v;
}

__global__ void bias_g_kernel(const float* __restrict__ a,
                              const float* __restrict__ b,
                              float* __restrict__ o) {
  int i = blockIdx.x * blockDim.x + threadIdx.x;
  if (i < G4) o[i] = a[i] + b[i];
}

// ------------------------- MLP GEMMs ---------------------------------------
// grid: (N/256, M/32), 256 threads = 8 waves, wave tile = 16(M) x 64(N).

__global__ void __launch_bounds__(256)
gemm1_kernel(const __bf16* __restrict__ A, const __bf16* __restrict__ W,
             const float* __restrict__ bias, __bf16* __restrict__ O) {
  const int lane = threadIdx.x & 31, w = threadIdx.x >> 5;
  const int mb = (blockIdx.y * 2 + (w >> 2)) * 16;
  const int nb = blockIdx.x * 256 + (w & 3) * 64;
  const int r16 = lane & 15, hi = lane >> 4, khalf = hi * 16;
  const __bf16* arow = A + (size_t)(mb + r16) * IND;
  v8f acc[4] = {{}, {}, {}, {}};
  for (int kt = 0; kt < IND; kt += 32) {
    v16bf a = load_a_frag(arow, kt, hi);
#pragma unroll
    for (int j = 0; j < 4; ++j) {
      v16bf b = load_b_frag(W + (size_t)(nb + j * 16 + r16) * IND, kt, khalf);
      acc[j] = wmma_bf16(a, b, acc[j]);
    }
  }
#pragma unroll
  for (int j = 0; j < 4; ++j) {
    int n = nb + j * 16 + r16;
    float bv = bias[n];
#pragma unroll
    for (int r = 0; r < 8; ++r) {
      int m = mb + r + 8 * hi;
      O[(size_t)m * MH + n] = (__bf16)(acc[j][r] + bv);
    }
  }
}

// GEMM2 + bias + ReLU, store time-major bf16: msb[(s*B + b)*MO + n]
__global__ void __launch_bounds__(256)
gemm2_kernel(const __bf16* __restrict__ A, const __bf16* __restrict__ W,
             const float* __restrict__ bias, __bf16* __restrict__ Msb) {
  const int lane = threadIdx.x & 31, w = threadIdx.x >> 5;
  const int mb = (blockIdx.y * 2 + (w >> 2)) * 16;
  const int nb = (w & 3) * 64;                       // N = 256, grid.x = 1
  const int r16 = lane & 15, hi = lane >> 4, khalf = hi * 16;
  const __bf16* arow = A + (size_t)(mb + r16) * MH;
  v8f acc[4] = {{}, {}, {}, {}};
  for (int kt = 0; kt < MH; kt += 32) {
    v16bf a = load_a_frag(arow, kt, hi);
#pragma unroll
    for (int j = 0; j < 4; ++j) {
      v16bf b = load_b_frag(W + (size_t)(nb + j * 16 + r16) * MH, kt, khalf);
      acc[j] = wmma_bf16(a, b, acc[j]);
    }
  }
#pragma unroll
  for (int j = 0; j < 4; ++j) {
    int n = nb + j * 16 + r16;
    float bv = bias[n];
#pragma unroll
    for (int r = 0; r < 8; ++r) {
      int m = mb + r + 8 * hi;
      int bb = m >> 7, ss = m & 127;                 // m = b*S + s
      float v = acc[j][r] + bv;
      v = v > 0.f ? v : 0.f;                         // ReLU (end of MLP)
      Msb[((size_t)ss * B_ + bb) * MO + n] = (__bf16)v;
    }
  }
}

// ------------------------- LSTM step kernels -------------------------------
// gates[256,4096] = [mlp_t | o_prev | h] @ Wg^T + bias, segmented K.
__global__ void __launch_bounds__(256)
gates_kernel(const __bf16* __restrict__ mlpt, const __bf16* __restrict__ oprev,
             const __bf16* __restrict__ hbf, const __bf16* __restrict__ Wg,
             const float* __restrict__ biasg, float* __restrict__ gates) {
  const int lane = threadIdx.x & 31, w = threadIdx.x >> 5;
  const int mb = (blockIdx.y * 2 + (w >> 2)) * 16;
  const int nb = blockIdx.x * 256 + (w & 3) * 64;
  const int r16 = lane & 15, hi = lane >> 4, khalf = hi * 16;
  const int arow = mb + r16;
  v8f acc[4] = {{}, {}, {}, {}};
  for (int kt = 0; kt < KG; kt += 32) {
    // Wave-uniform segment select (kt is uniform; EXEC stays all-ones).
    const __bf16* ap; int ko;
    if (kt < MO)        { ap = mlpt  + (size_t)arow * MO;  ko = kt;       }
    else if (kt < LIN)  { ap = oprev + (size_t)arow * ACT; ko = kt - MO;  }
    else                { ap = hbf   + (size_t)arow * HID; ko = kt - LIN; }
    v16bf a = load_a_frag(ap, ko, hi);
    if (kt + 32 < KG)   // warm L2->L0 path for next weight tile
      __builtin_prefetch(Wg + (size_t)(nb + r16) * KG + kt + 32, 0, 1);
#pragma unroll
    for (int j = 0; j < 4; ++j) {
      v16bf b = load_b_frag(Wg + (size_t)(nb + j * 16 + r16) * KG, kt, khalf);
      acc[j] = wmma_bf16(a, b, acc[j]);
    }
  }
#pragma unroll
  for (int j = 0; j < 4; ++j) {
    int n = nb + j * 16 + r16;
    float bv = biasg[n];
#pragma unroll
    for (int r = 0; r < 8; ++r) {
      int m = mb + r + 8 * hi;
      gates[(size_t)m * G4 + n] = acc[j][r] + bv;
    }
  }
}

__global__ void cell_kernel(const float* __restrict__ gates,
                            float* __restrict__ c, __bf16* __restrict__ hbf) {
  int idx = blockIdx.x * blockDim.x + threadIdx.x;
  if (idx >= B_ * HID) return;
  int bb = idx >> 10, j = idx & 1023;
  const float* g = gates + (size_t)bb * G4;
  float ig = 1.f / (1.f + __expf(-g[j]));
  float fg = 1.f / (1.f + __expf(-g[j + 1024]));
  float gg = tanhf(g[j + 2048]);
  float og = 1.f / (1.f + __expf(-g[j + 3072]));
  float cv = fg * c[idx] + ig * gg;
  c[idx] = cv;
  hbf[idx] = (__bf16)(og * tanhf(cv));
}

// out = h @ W_out^T + b_out : f32 to d_out[:,t,:] and bf16 to o_prev buffer.
// grid: B/32 blocks; waves 0-3 -> m_tile even (n = w*16), 4-7 -> m_tile odd.
__global__ void __launch_bounds__(256)
outproj_kernel(const __bf16* __restrict__ hbf, const __bf16* __restrict__ Wo,
               const float* __restrict__ bo, float* __restrict__ out_t,
               __bf16* __restrict__ oprev) {
  const int lane = threadIdx.x & 31, w = threadIdx.x >> 5;
  const int mb = (blockIdx.x * 2 + (w >> 2)) * 16;
  const int nb = (w & 3) * 16;                        // N = 64
  const int r16 = lane & 15, hi = lane >> 4, khalf = hi * 16;
  const __bf16* arow = hbf + (size_t)(mb + r16) * HID;
  const __bf16* brow = Wo + (size_t)(nb + r16) * HID;
  v8f acc = {};
  for (int kt = 0; kt < HID; kt += 32) {
    v16bf a = load_a_frag(arow, kt, hi);
    v16bf b = load_b_frag(brow, kt, khalf);
    acc = wmma_bf16(a, b, acc);
  }
  int n = nb + r16;
  float bv = bo[n];
#pragma unroll
  for (int r = 0; r < 8; ++r) {
    int m = mb + r + 8 * hi;                          // batch index
    float v = acc[r] + bv;
    out_t[(size_t)m * (S_ * ACT) + n] = v;            // d_out[b, t, n]
    oprev[m * ACT + n] = (__bf16)v;                   // feeds step t+1
  }
}

// ------------------------- launcher ----------------------------------------

extern "C" void kernel_launch(void* const* d_in, const int* in_sizes, int n_in,
                              void* d_out, int out_size, void* d_ws, size_t ws_size,
                              hipStream_t stream) {
  (void)in_sizes; (void)n_in; (void)out_size; (void)ws_size;
  const float* obs   = (const float*)d_in[0];
  const float* obst  = (const float*)d_in[1];
  const float* msk   = (const float*)d_in[2];
  const float* W1    = (const float*)d_in[3];
  const float* b1    = (const float*)d_in[4];
  const float* W2    = (const float*)d_in[5];
  const float* b2    = (const float*)d_in[6];
  const float* W_ih  = (const float*)d_in[7];
  const float* b_ih  = (const float*)d_in[8];
  const float* W_hh  = (const float*)d_in[9];
  const float* b_hh  = (const float*)d_in[10];
  const float* W_out = (const float*)d_in[11];
  const float* b_out = (const float*)d_in[12];
  float* out = (float*)d_out;

  uint8_t* base = (uint8_t*)d_ws;
  size_t off = 0;
  auto alloc = [&](size_t bytes) {
    void* r = base + off;
    off = (off + bytes + 255) & ~(size_t)255;
    return r;
  };
  __bf16* xbf  = (__bf16*)alloc((size_t)B_ * S_ * IND * 2);
  __bf16* W1b  = (__bf16*)alloc((size_t)MH * IND * 2);
  __bf16* h1b  = (__bf16*)alloc((size_t)B_ * S_ * MH * 2);
  __bf16* W2b  = (__bf16*)alloc((size_t)MO * MH * 2);
  __bf16* msb  = (__bf16*)alloc((size_t)B_ * S_ * MO * 2);   // [S,B,MO]
  __bf16* Wgb  = (__bf16*)alloc((size_t)G4 * KG * 2);
  float*  bg   = (float*) alloc((size_t)G4 * 4);
  __bf16* Wob  = (__bf16*)alloc((size_t)ACT * HID * 2);
  float*  gts  = (float*) alloc((size_t)B_ * G4 * 4);
  float*  cbuf = (float*) alloc((size_t)B_ * HID * 4);
  __bf16* hbf  = (__bf16*)alloc((size_t)B_ * HID * 2);
  __bf16* opv  = (__bf16*)alloc((size_t)B_ * ACT * 2);

  // h0 = c0 = o_prev0 = 0 (bf16 zero bits == 0.0)
  hipMemsetAsync(cbuf, 0, (size_t)B_ * HID * 4, stream);
  hipMemsetAsync(hbf,  0, (size_t)B_ * HID * 2, stream);
  hipMemsetAsync(opv,  0, (size_t)B_ * ACT * 2, stream);

  const int T = 256;
  pack_x_kernel<<<(B_ * S_ * IND + T - 1) / T, T, 0, stream>>>(obs, obst, msk, xbf);
  cvt_bf16_kernel<<<(MH * IND + T - 1) / T, T, 0, stream>>>(W1, W1b, MH * IND);
  cvt_bf16_kernel<<<(MO * MH + T - 1) / T, T, 0, stream>>>(W2, W2b, MO * MH);
  cvt_bf16_kernel<<<(ACT * HID + T - 1) / T, T, 0, stream>>>(W_out, Wob, ACT * HID);
  pack_wg_kernel<<<(G4 * KG + T - 1) / T, T, 0, stream>>>(W_ih, W_hh, Wgb);
  bias_g_kernel<<<(G4 + T - 1) / T, T, 0, stream>>>(b_ih, b_hh, bg);

  gemm1_kernel<<<dim3(MH / 256, (B_ * S_) / 32), 256, 0, stream>>>(xbf, W1b, b1, h1b);
  gemm2_kernel<<<dim3(1, (B_ * S_) / 32), 256, 0, stream>>>(h1b, W2b, b2, msb);

  for (int t = 0; t < S_; ++t) {
    gates_kernel<<<dim3(G4 / 256, B_ / 32), 256, 0, stream>>>(
        msb + (size_t)t * B_ * MO, opv, hbf, Wgb, bg, gts);
    cell_kernel<<<(B_ * HID + T - 1) / T, T, 0, stream>>>(gts, cbuf, hbf);
    outproj_kernel<<<B_ / 32, 256, 0, stream>>>(
        hbf, Wob, b_out, out + (size_t)t * ACT, opv);
  }
}
